// Conformer_85212151152801
// MI455X (gfx1250) — compile-verified, matrix operand order checked by reading, not run on hardware
//
#include <hip/hip_runtime.h>
#include <hip/hip_bf16.h>
#include <stdint.h>

// ---------------------------------------------------------------------------
// Conformer forward for gfx1250 (MI455X).
//  * All matmuls: v_wmma_f32_16x16x32_bf16, fp32 accumulate.
//  * 64x64 block tile, 8 waves, 16x32 wave tile (2 WMMA/wave/k-step, A reuse).
//  * Fragment loads = ds_load_b128 pairs; staging = global_load_b128 pairs +
//    v_cvt_pk_bf16_f32 + ds_store_b128; double-buffered LDS pipeline.
//  * conv2 implicit GEMM: pure-bf16 tiles moved with global_load_async_to_lds_b128
//    (ASYNCcnt) overlapping WMMA, s_wait_asynccnt sync.
// ---------------------------------------------------------------------------

typedef __bf16 v16bf __attribute__((ext_vector_type(16)));
typedef __bf16 v8bf  __attribute__((ext_vector_type(8)));
typedef float  v8f   __attribute__((ext_vector_type(8)));

#define LDSP 40   // LDS row stride in bf16 elems: 80B per row (16B-aligned, conflict-skewed)

__device__ __forceinline__ float dev_sigmoid(float v) { return 1.f / (1.f + __expf(-v)); }
__device__ __forceinline__ float dev_silu(float v)    { return v * dev_sigmoid(v); }

// 8 contiguous fp32 -> v8bf (2x b128 load + packed converts)
__device__ __forceinline__ v8bf cvt8(const float* __restrict__ s)
{
    float4 f0 = *(const float4*)s;
    float4 f1 = *(const float4*)(s + 4);
    v8bf o;
    o[0] = (__bf16)f0.x; o[1] = (__bf16)f0.y; o[2] = (__bf16)f0.z; o[3] = (__bf16)f0.w;
    o[4] = (__bf16)f1.x; o[5] = (__bf16)f1.y; o[6] = (__bf16)f1.z; o[7] = (__bf16)f1.w;
    return o;
}

#define FRAG16(p) __builtin_shufflevector((p)[0], (p)[2], 0,1,2,3,4,5,6,7,8,9,10,11,12,13,14,15)

// ---------------------------------------------------------------------------
// General strided / batched GEMM:  C = act( (alpha * A@B^T + bias) * gamma )
//   A[m,k] = A + m*sAm + k          (K-contiguous; K % 32 == 0)
//   B[n,k] = B + n*sBn + k          (K-contiguous)
//   C[m,n] = C + m*sCm + n
// 256 threads = 8 waves (4m x 2n); block tile 64x64; wave tile 16x32; K-step 32.
// ---------------------------------------------------------------------------
__global__ __launch_bounds__(256) void k_gemm(
    const float* __restrict__ A, const float* __restrict__ B,
    float* __restrict__ C, const float* __restrict__ bias,
    int M, int N, int K,
    long long sAm, long long sBn, long long sCm,
    long long bAb, long long bAh, long long bBb, long long bBh,
    long long bCb, long long bCh, int nh,
    float alpha, float gamma, int act)
{
    __shared__ __bf16 sA[2][64 * LDSP];
    __shared__ __bf16 sB[2][64 * LDSP];    // transposed: [n][k]

    int z  = blockIdx.z;
    int zb = z / nh, zh = z % nh;
    A += zb * bAb + zh * bAh;
    B += zb * bBb + zh * bBh;
    C += zb * bCb + zh * bCh;

    int m0 = blockIdx.y * 64;
    int n0 = blockIdx.x * 64;
    int tid  = threadIdx.x;
    int lane = tid & 31, wave = tid >> 5;
    int wm = wave & 3, wn = wave >> 2;         // 4 x 2 wave grid
    int half = lane >> 4, l16 = lane & 15;

    // staging ownership: each thread owns 8 contiguous K of one A row and one B row
    int arow = tid >> 2;            // 0..63
    int akc  = (tid & 3) << 3;      // 0,8,16,24

    bool aok = (m0 + arow) < M;
    bool bok = (n0 + arow) < N;
    const float* Arow = A + (long long)(m0 + arow) * sAm + akc;
    const float* Brow = B + (long long)(n0 + arow) * sBn + akc;

    auto stage = [&](int buf, int kk0) {
        v8bf av = {};
        if (aok) av = cvt8(Arow + kk0);
        *(v8bf*)&sA[buf][arow * LDSP + akc] = av;
        v8bf bv = {};
        if (bok) bv = cvt8(Brow + kk0);
        *(v8bf*)&sB[buf][arow * LDSP + akc] = bv;
    };

    v8f acc0 = {}, acc1 = {};
    int nit = K >> 5;

    stage(0, 0);
    __syncthreads();

    for (int it = 0; it < nit; ++it) {
        int cur = it & 1;
        if (it + 1 < nit) {
            stage(cur ^ 1, (it + 1) << 5);
            if (it + 2 < nit && aok)
                __builtin_prefetch(Arow + ((it + 2) << 5), 0, 0);   // global_prefetch_b8
        }
        const v8bf* pa  = (const v8bf*)&sA[cur][(wm * 16 + l16) * LDSP + 8 * half];
        const v8bf* pb0 = (const v8bf*)&sB[cur][(wn * 32 + l16) * LDSP + 8 * half];
        const v8bf* pb1 = (const v8bf*)&sB[cur][(wn * 32 + 16 + l16) * LDSP + 8 * half];
        v16bf af  = FRAG16(pa);
        v16bf bf0 = FRAG16(pb0);
        v16bf bf1 = FRAG16(pb1);
        acc0 = __builtin_amdgcn_wmma_f32_16x16x32_bf16(false, af, false, bf0,
                                                       (short)0, acc0, false, false);
        acc1 = __builtin_amdgcn_wmma_f32_16x16x32_bf16(false, af, false, bf1,
                                                       (short)0, acc1, false, false);
        __syncthreads();
    }

    // C/D layout: VGPR r -> M = r + 8*half, N = l16
    #pragma unroll
    for (int sub = 0; sub < 2; ++sub) {
        v8f acc = sub ? acc1 : acc0;
        #pragma unroll
        for (int r = 0; r < 8; ++r) {
            int mm = m0 + wm * 16 + r + 8 * half;
            int nn = n0 + wn * 32 + sub * 16 + l16;
            if (mm < M && nn < N) {
                float v = acc[r] * alpha;
                if (bias) v += bias[nn];
                v *= gamma;
                if (act == 1)      v = dev_silu(v);
                else if (act == 2) v = fmaxf(v, 0.f);
                C[(long long)mm * sCm + nn] = v;
            }
        }
    }
}

// ---------------------------------------------------------------------------
// Subsampling conv #1: x(8,1024,80) -> NHWC bf16 (8, 512h, 40w, 512c),
// stride2 pad1, bias+ReLU. co fastest (coalesced; NHWC makes conv2 im2col
// K-chunks contiguous 16B runs for async DMA).
// ---------------------------------------------------------------------------
__global__ __launch_bounds__(256) void k_conv1(
    const float* __restrict__ x, const float* __restrict__ w,
    const float* __restrict__ b, __bf16* __restrict__ out)
{
    long long i = (long long)blockIdx.x * 256 + threadIdx.x;   // 8*512*40*512
    int co = (int)(i & 511);
    long long j = i >> 9;
    int tw = (int)(j % 40); j /= 40;
    int th = (int)(j % 512);
    int bb = (int)(j / 512);
    float acc = b[co];
    #pragma unroll
    for (int kh = 0; kh < 3; ++kh) {
        int hi = 2 * th + kh - 1;
        if ((unsigned)hi >= 1024u) continue;
        #pragma unroll
        for (int kw = 0; kw < 3; ++kw) {
            int wi = 2 * tw + kw - 1;
            if ((unsigned)wi >= 80u) continue;
            acc += x[((long long)bb * 1024 + hi) * 80 + wi] * w[co * 9 + kh * 3 + kw];
        }
    }
    out[i] = (__bf16)fmaxf(acc, 0.f);
}

// pack sub_w2 (512,512,3,3) -> bf16 (512n, 4608k), k = (kh*3+kw)*512 + ci
__global__ __launch_bounds__(256) void k_pack_w2(const float* __restrict__ w,
                                                 __bf16* __restrict__ o)
{
    long long i = (long long)blockIdx.x * 256 + threadIdx.x;   // 512*4608
    int k = (int)(i % 4608);
    int n = (int)(i / 4608);
    int tap = k >> 9, ci = k & 511;
    o[i] = (__bf16)w[((long long)n * 512 + ci) * 9 + tap];
}

// ---------------------------------------------------------------------------
// Subsampling conv #2 as implicit GEMM (WMMA): M=40960 (b,th,tw), N=512, K=4608.
// Raw bf16 tiles moved with global_load_async_to_lds_b128 (ASYNCcnt), overlapping
// WMMA of the previous tile; halo chunks zero-filled via ds_store_b128.
// Fuses bias + ReLU + permute: writes A_proj[(b*256+th)*10240 + n*20 + tw].
// ---------------------------------------------------------------------------
__global__ __launch_bounds__(256) void k_conv2_wmma(
    const __bf16* __restrict__ in,   // NHWC (8, 512h, 40w, 512ci) bf16
    const __bf16* __restrict__ wp,   // (512n, 4608k) bf16 packed
    const float* __restrict__ bias,  // sub_b2
    float* __restrict__ out)         // (2048, 10240)
{
    __shared__ __bf16 sA[2][64 * LDSP];
    __shared__ __bf16 sB[2][64 * LDSP];
    const int K = 4608;
    int m0 = blockIdx.y * 64;
    int n0 = blockIdx.x * 64;
    int tid  = threadIdx.x;
    int lane = tid & 31, wave = tid >> 5;
    int wm = wave & 3, wn = wave >> 2;
    int half = lane >> 4, l16 = lane & 15;

    int arow = tid >> 2;            // tile row for both A(m) and B(n)
    int akc  = (tid & 3) << 3;      // k chunk base

    int m = m0 + arow;
    int b  = m / 5120, rr = m % 5120;
    int th = rr / 20,  tw = rr % 20;
    const __bf16* Brow = wp + (long long)(n0 + arow) * 4608 + akc;

    auto stage = [&](int buf, int kk0) {
        // ---- A (im2col gather, one 16B chunk per thread, async DMA) ----
        int k   = kk0 + akc;
        int tap = k >> 9, ci = k & 511;          // chunk never crosses a tap
        int kh = tap / 3, kw = tap % 3;
        int hi = 2 * th + kh - 1, wi = 2 * tw + kw - 1;
        __bf16* adst = &sA[buf][arow * LDSP + akc];
        if ((unsigned)hi < 512u && (unsigned)wi < 40u) {
            const __bf16* asrc = in + (((long long)b * 512 + hi) * 40 + wi) * 512 + ci;
            asm volatile("global_load_async_to_lds_b128 %0, %1, off"
                         :: "v"((uint32_t)(uintptr_t)adst),
                            "v"((uint64_t)(uintptr_t)asrc)
                         : "memory");
        } else {
            *(v8bf*)adst = v8bf{};               // zero halo
        }
        // ---- B (packed weights, raw async DMA) ----
        __bf16* bdst = &sB[buf][arow * LDSP + akc];
        asm volatile("global_load_async_to_lds_b128 %0, %1, off"
                     :: "v"((uint32_t)(uintptr_t)bdst),
                        "v"((uint64_t)(uintptr_t)(Brow + kk0))
                     : "memory");
    };

    v8f acc0 = {}, acc1 = {};
    const int nit = K >> 5;

    stage(0, 0);
    asm volatile("s_wait_asynccnt 0x0" ::: "memory");
    __syncthreads();

    for (int it = 0; it < nit; ++it) {
        int cur = it & 1;
        if (it + 1 < nit) stage(cur ^ 1, (it + 1) << 5);
        const v8bf* pa  = (const v8bf*)&sA[cur][(wm * 16 + l16) * LDSP + 8 * half];
        const v8bf* pb0 = (const v8bf*)&sB[cur][(wn * 32 + l16) * LDSP + 8 * half];
        const v8bf* pb1 = (const v8bf*)&sB[cur][(wn * 32 + 16 + l16) * LDSP + 8 * half];
        v16bf af  = FRAG16(pa);
        v16bf bf0 = FRAG16(pb0);
        v16bf bf1 = FRAG16(pb1);
        acc0 = __builtin_amdgcn_wmma_f32_16x16x32_bf16(false, af, false, bf0,
                                                       (short)0, acc0, false, false);
        acc1 = __builtin_amdgcn_wmma_f32_16x16x32_bf16(false, af, false, bf1,
                                                       (short)0, acc1, false, false);
        if (it + 1 < nit)
            asm volatile("s_wait_asynccnt 0x0" ::: "memory");
        __syncthreads();
    }

    #pragma unroll
    for (int sub = 0; sub < 2; ++sub) {
        v8f acc = sub ? acc1 : acc0;
        #pragma unroll
        for (int r = 0; r < 8; ++r) {
            int mm = m0 + wm * 16 + r + 8 * half;
            int nn = n0 + wn * 32 + sub * 16 + l16;
            int ob = mm / 5120, orr = mm % 5120;
            int oth = orr / 20, otw = orr % 20;
            float v = fmaxf(acc[r] + bias[nn], 0.f);
            out[(long long)(ob * 256 + oth) * 10240 + nn * 20 + otw] = v;
        }
    }
}

// ---------------------------------------------------------------------------
// LayerNorm over D=512, block per row (256 threads, 2 elems/thread). In-place safe.
// ---------------------------------------------------------------------------
__global__ __launch_bounds__(256) void k_ln(const float* __restrict__ x,
                                            float* __restrict__ y,
                                            const float* __restrict__ g,
                                            const float* __restrict__ b)
{
    long long row = blockIdx.x;
    const float* xr = x + row * 512;
    float* yr = y + row * 512;
    int t = threadIdx.x;
    float a0 = xr[t], a1 = xr[t + 256];
    __shared__ float sm[256];
    sm[t] = a0 + a1; __syncthreads();
    for (int off = 128; off > 0; off >>= 1) { if (t < off) sm[t] += sm[t + off]; __syncthreads(); }
    float mean = sm[0] * (1.f / 512.f);
    __syncthreads();
    float d0 = a0 - mean, d1 = a1 - mean;
    sm[t] = d0 * d0 + d1 * d1; __syncthreads();
    for (int off = 128; off > 0; off >>= 1) { if (t < off) sm[t] += sm[t + off]; __syncthreads(); }
    float r = rsqrtf(sm[0] * (1.f / 512.f) + 1e-5f);
    yr[t]       = d0 * r * g[t]       + b[t];
    yr[t + 256] = d1 * r * g[t + 256] + b[t + 256];
}

// x += a*y
__global__ __launch_bounds__(256) void k_axpy(float* __restrict__ x,
                                              const float* __restrict__ y, float a)
{
    long long i = (long long)blockIdx.x * 256 + threadIdx.x;
    x[i] += a * y[i];
}

// qu/qv = q + pos_u / pos_v  (broadcast over (b,t); layout (b,t,h,d))
__global__ __launch_bounds__(256) void k_addpos(const float* __restrict__ q,
                                                const float* __restrict__ pu,
                                                const float* __restrict__ pv,
                                                float* __restrict__ qu,
                                                float* __restrict__ qv)
{
    long long i = (long long)blockIdx.x * 256 + threadIdx.x;
    int hd = (int)(i & 511);
    float v = q[i];
    qu[i] = v + pu[hd];
    qv[i] = v + pv[hd];
}

// vT[(b,h,d),t] = v[(b,t),(h,d)]  : makes attn@v B-operand K-contiguous
__global__ __launch_bounds__(256) void k_transpose_v(const float* __restrict__ v,
                                                     float* __restrict__ vt)
{
    long long i = (long long)blockIdx.x * 256 + threadIdx.x;   // 8*8*64*256
    int t = (int)(i & 255);
    int d = (int)((i >> 8) & 63);
    int h = (int)((i >> 14) & 7);
    int b = (int)(i >> 17);
    vt[i] = v[((long long)(b * 256 + t)) * 512 + h * 64 + d];
}

// GLU: z(2048,1024) -> o(2048,512) = z[:, :512] * sigmoid(z[:, 512:])
__global__ __launch_bounds__(256) void k_glu(const float* __restrict__ z,
                                             float* __restrict__ o)
{
    long long i = (long long)blockIdx.x * 256 + threadIdx.x;
    long long row = i >> 9; int c = (int)(i & 511);
    float a = z[(row << 10) + c];
    float g = z[(row << 10) + 512 + c];
    o[i] = a * dev_sigmoid(g);
}

// depthwise conv(k=31, SAME) over t + bias + BN-affine + SiLU. y,(b,t,c) rows.
__global__ __launch_bounds__(256) void k_dwconv(const float* __restrict__ y,
                                                float* __restrict__ o,
                                                const float* __restrict__ w,
                                                const float* __restrict__ bias,
                                                const float* __restrict__ bng,
                                                const float* __restrict__ bnb)
{
    long long i = (long long)blockIdx.x * 256 + threadIdx.x;   // 8*256*512
    int c = (int)(i & 511);
    int t = (int)((i >> 9) & 255);
    int b = (int)(i >> 17);
    float acc = bias[c];
    const float* wc = w + c * 31;
    const float* base = y + ((long long)b << 17) + c;
    #pragma unroll
    for (int j = 0; j < 31; ++j) {
        int tt = t + j - 15;
        if ((unsigned)tt < 256u) acc += base[(long long)tt << 9] * wc[j];
    }
    const float bnscale = 0.99999500003749969f;   // 1/sqrt(1+1e-5)
    acc = acc * bnscale * bng[c] + bnb[c];
    o[i] = dev_silu(acc);
}

// softmax over s with fused rel_shift:  val = sc[t,s] + bd[t, 255 - t + s]
__global__ __launch_bounds__(256) void k_softmax_rel(const float* __restrict__ sc,
                                                     const float* __restrict__ bd,
                                                     float* __restrict__ attn)
{
    long long rid = blockIdx.x;          // (b*8+h)*256 + t
    int t = (int)(rid & 255);
    long long bh = rid >> 8;
    const float* srow = sc + rid * 256;
    const float* brow = bd + (bh * 256 + t) * 511;
    int s = threadIdx.x;
    float v = srow[s] + brow[255 - t + s];
    __shared__ float sm[256];
    sm[s] = v; __syncthreads();
    for (int off = 128; off > 0; off >>= 1) { if (s < off) sm[s] = fmaxf(sm[s], sm[s + off]); __syncthreads(); }
    float mx = sm[0]; __syncthreads();
    float e = __expf(v - mx);
    sm[s] = e; __syncthreads();
    for (int off = 128; off > 0; off >>= 1) { if (s < off) sm[s] += sm[s + off]; __syncthreads(); }
    attn[rid * 256 + s] = e / sm[0];
}

// relative positional encodings: 511 x 512 (pos = 255 - row)
__global__ __launch_bounds__(256) void k_pe(float* __restrict__ pe)
{
    long long i = (long long)blockIdx.x * 256 + threadIdx.x;
    if (i >= 511LL * 512) return;
    int col = (int)(i & 511);
    int row = (int)(i >> 9);
    float pos = 255.f - (float)row;
    float div = __expf((float)(col & ~1) * (-9.210340371976184f / 512.f));
    float ang = pos * div;
    pe[i] = (col & 1) ? __cosf(ang) : __sinf(ang);
}

// ---------------------------------------------------------------------------
// Host orchestration
// ---------------------------------------------------------------------------
extern "C" void kernel_launch(void* const* d_in, const int* in_sizes, int n_in,
                              void* d_out, int out_size, void* d_ws, size_t ws_size,
                              hipStream_t stream)
{
    (void)in_sizes; (void)n_in; (void)out_size; (void)ws_size;
    // Input order = JAX pytree flatten (dict keys sorted):
    // params.blocks.{sorted leaf names}[0..36], dec_b=37, dec_w=38, proj_b=39,
    // proj_w=40, sub_b1=41, sub_b2=42, sub_w1=43, sub_w2=44, x=45.
    auto in = [&](int i) { return (const float*)d_in[i]; };

    const int  B = 8, T = 256, D = 512, H = 2048, NH = 8, DK = 64;
    const int  ROWS = B * T;                 // 2048
    const long long RD = (long long)ROWS * D;

    // ---- workspace arena ----
    char* wsb = (char*)d_ws;
    size_t off = 0;
    auto alloc = [&](size_t bytes) -> void* {
        void* p = wsb + off;
        off += (bytes + 255) & ~(size_t)255;
        return p;
    };
    __bf16* conv1out = (__bf16*)alloc((size_t)8 * 512 * 40 * 512 * sizeof(__bf16)); // NHWC, 167.8 MB
    float*  A_proj   = (float*)alloc((size_t)ROWS * 10240 * sizeof(float));          // 83.9 MB
    __bf16* w2p      = (__bf16*)alloc((size_t)512 * 4608 * sizeof(__bf16));
    float*  posemb   = (float*)alloc((size_t)511 * 512 * sizeof(float));
    float*  hbuf     = (float*)alloc((size_t)RD * sizeof(float));

    // per-block temporaries alias the (dead after projection) conv1out region
    size_t toff = 0;
    auto talloc = [&](size_t nfloat) -> float* {
        float* p = (float*)(wsb + toff);
        toff += (nfloat * sizeof(float) + 255) & ~(size_t)255;
        return p;
    };
    float* y    = talloc(RD);
    float* t1   = talloc((size_t)ROWS * H);
    float* t2   = talloc(RD);
    float* qb   = talloc(RD);
    float* kb   = talloc(RD);
    float* vb   = talloc(RD);
    float* qu   = talloc(RD);
    float* qv   = talloc(RD);
    float* vT   = talloc(RD);
    float* pbuf = talloc((size_t)511 * 512);
    float* bdb  = talloc((size_t)B * NH * T * 511);
    float* scb  = talloc((size_t)B * NH * T * T);
    float* atb  = talloc((size_t)B * NH * T * T);
    float* ob   = talloc(RD);
    float* y2   = talloc(RD);
    float* y3   = talloc(RD);

    auto gemm = [&](const float* A, long long sAm,
                    const float* Bm, long long sBn,
                    float* C, long long sCm, const float* bias,
                    int M, int N, int K, float alpha, float gamma, int act,
                    int nbatch = 1, int nh = 1,
                    long long bAb = 0, long long bAh = 0,
                    long long bBb = 0, long long bBh = 0,
                    long long bCb = 0, long long bCh = 0) {
        dim3 g((N + 63) / 64, (M + 63) / 64, nbatch);
        k_gemm<<<g, 256, 0, stream>>>(A, Bm, C, bias, M, N, K,
                                      sAm, sBn, sCm,
                                      bAb, bAh, bBb, bBh, bCb, bCh, nh,
                                      alpha, gamma, act);
    };

    // ---- subsampling + projection ----
    k_conv1<<<(8LL * 512 * 40 * 512) / 256, 256, 0, stream>>>(in(45), in(43), in(41), conv1out);
    k_pack_w2<<<(512LL * 4608) / 256, 256, 0, stream>>>(in(44), w2p);
    {
        dim3 g(512 / 64, 40960 / 64, 1);
        k_conv2_wmma<<<g, 256, 0, stream>>>(conv1out, w2p, in(42), A_proj);
    }
    // h = (A_proj @ proj_w^T + proj_b) * sqrt(512)
    gemm(A_proj, 10240, in(40), 10240, hbuf, 512, in(39),
         ROWS, 512, 10240, 1.f, 22.627416997969522f, 0);
    k_pe<<<(511 * 512 + 255) / 256, 256, 0, stream>>>(posemb);

    // ---- conformer blocks ----
    for (int blk = 0; blk < 8; ++blk) {
        long long b512 = (long long)blk * 512;
        const float* attn_ln_b = in(0)  + b512;
        const float* attn_ln_g = in(1)  + b512;
        const float* bk_       = in(2)  + b512;
        const float* bn_b      = in(3)  + b512;
        const float* bn_g      = in(4)  + b512;
        const float* bo_       = in(5)  + b512;
        const float* bq_       = in(6)  + b512;
        const float* bv_       = in(7)  + b512;
        const float* conv_ln_b = in(8)  + b512;
        const float* conv_ln_g = in(9)  + b512;
        const float* dw_b      = in(10) + b512;
        const float* dw_w      = in(11) + (long long)blk * 512 * 31;
        const float* ff1_b1    = in(12) + (long long)blk * 2048;
        const float* ff1_b2    = in(13) + b512;
        const float* ff1_ln_b  = in(14) + b512;
        const float* ff1_ln_g  = in(15) + b512;
        const float* ff1_w1    = in(16) + (long long)blk * 2048 * 512;
        const float* ff1_w2    = in(17) + (long long)blk * 512 * 2048;
        const float* ff2_b1    = in(18) + (long long)blk * 2048;
        const float* ff2_b2    = in(19) + b512;
        const float* ff2_ln_b  = in(20) + b512;
        const float* ff2_ln_g  = in(21) + b512;
        const float* ff2_w1    = in(22) + (long long)blk * 2048 * 512;
        const float* ff2_w2    = in(23) + (long long)blk * 512 * 2048;
        const float* out_ln_b  = in(24) + b512;
        const float* out_ln_g  = in(25) + b512;
        const float* pos_u     = in(26) + b512;
        const float* pos_v     = in(27) + b512;
        const float* pw1_b     = in(28) + (long long)blk * 1024;
        const float* pw1_w     = in(29) + (long long)blk * 1024 * 512;
        const float* pw2_b     = in(30) + b512;
        const float* pw2_w     = in(31) + (long long)blk * 512 * 512;
        const float* wk_       = in(32) + (long long)blk * 512 * 512;
        const float* wo_       = in(33) + (long long)blk * 512 * 512;
        const float* wpos      = in(34) + (long long)blk * 512 * 512;
        const float* wq_       = in(35) + (long long)blk * 512 * 512;
        const float* wv_       = in(36) + (long long)blk * 512 * 512;

        // --- FF1 (half-step) ---
        k_ln<<<ROWS, 256, 0, stream>>>(hbuf, y, ff1_ln_g, ff1_ln_b);
        gemm(y, 512, ff1_w1, 512, t1, 2048, ff1_b1, ROWS, H, D, 1.f, 1.f, 1);
        gemm(t1, 2048, ff1_w2, 2048, t2, 512, ff1_b2, ROWS, D, H, 1.f, 1.f, 0);
        k_axpy<<<RD / 256, 256, 0, stream>>>(hbuf, t2, 0.5f);

        // --- MHSA with relative positions ---
        k_ln<<<ROWS, 256, 0, stream>>>(hbuf, y, attn_ln_g, attn_ln_b);
        gemm(y, 512, wq_, 512, qb, 512, bq_, ROWS, D, D, 1.f, 1.f, 0);
        gemm(y, 512, wk_, 512, kb, 512, bk_, ROWS, D, D, 1.f, 1.f, 0);
        gemm(y, 512, wv_, 512, vb, 512, bv_, ROWS, D, D, 1.f, 1.f, 0);
        gemm(posemb, 512, wpos, 512, pbuf, 512, nullptr, 511, D, D, 1.f, 1.f, 0);
        k_addpos<<<RD / 256, 256, 0, stream>>>(qb, pos_u, pos_v, qu, qv);
        k_transpose_v<<<RD / 256, 256, 0, stream>>>(vb, vT);
        // bd[b,h,t,p] = (1/8) qv . p     (batch over b*h)
        gemm(qv, 512, pbuf, 512, bdb, 511, nullptr, T, 511, DK, 0.125f, 1.f, 0,
             B * NH, NH, (long long)T * 512, 64, 0, 64,
             (long long)NH * T * 511, (long long)T * 511);
        // scores[b,h,t,s] = (1/8) qu . k
        gemm(qu, 512, kb, 512, scb, 256, nullptr, T, T, DK, 0.125f, 1.f, 0,
             B * NH, NH, (long long)T * 512, 64, (long long)T * 512, 64,
             (long long)NH * T * T, (long long)T * T);
        k_softmax_rel<<<B * NH * T, 256, 0, stream>>>(scb, bdb, atb);
        // o[b,t,h,d] = attn @ vT    (vT is (b,h,d,t), K-contiguous)
        gemm(atb, 256, vT, 256, ob, 512, nullptr, T, DK, T, 1.f, 1.f, 0,
             B * NH, NH, (long long)NH * T * T, (long long)T * T,
             (long long)NH * DK * T, (long long)DK * T,
             (long long)T * 512, 64);
        gemm(ob, 512, wo_, 512, t2, 512, bo_, ROWS, D, D, 1.f, 1.f, 0);
        k_axpy<<<RD / 256, 256, 0, stream>>>(hbuf, t2, 1.f);

        // --- Convolution module ---
        k_ln<<<ROWS, 256, 0, stream>>>(hbuf, y, conv_ln_g, conv_ln_b);
        gemm(y, 512, pw1_w, 512, t1, 1024, pw1_b, ROWS, 1024, D, 1.f, 1.f, 0);
        k_glu<<<RD / 256, 256, 0, stream>>>(t1, y2);
        k_dwconv<<<RD / 256, 256, 0, stream>>>(y2, y3, dw_w, dw_b, bn_g, bn_b);
        gemm(y3, 512, pw2_w, 512, t2, 512, pw2_b, ROWS, D, D, 1.f, 1.f, 0);
        k_axpy<<<RD / 256, 256, 0, stream>>>(hbuf, t2, 1.f);

        // --- FF2 (half-step) + out LN ---
        k_ln<<<ROWS, 256, 0, stream>>>(hbuf, y, ff2_ln_g, ff2_ln_b);
        gemm(y, 512, ff2_w1, 512, t1, 2048, ff2_b1, ROWS, H, D, 1.f, 1.f, 1);
        gemm(t1, 2048, ff2_w2, 2048, t2, 512, ff2_b2, ROWS, D, H, 1.f, 1.f, 0);
        k_axpy<<<RD / 256, 256, 0, stream>>>(hbuf, t2, 0.5f);
        k_ln<<<ROWS, 256, 0, stream>>>(hbuf, hbuf, out_ln_g, out_ln_b);   // in-place safe
    }

    // ---- decoder ----
    gemm(hbuf, 512, in(38), 512, (float*)d_out, 128, in(37),
         ROWS, 128, D, 1.f, 1.f, 0);
}
